// CombinedLoss_18562848654045
// MI455X (gfx1250) — matrix operand order repaired and checked by reading, not compiled, not run on hardware
//
#include <hip/hip_runtime.h>

// ---------------------------------------------------------------------------
// CombinedLoss: 0.5*mean|o-t| + 0.5*mean|CWT(o)-CWT(t)|  (real Morlet, 36 widths)
// Linearity: CWT(o)-CWT(t) = CWT(o-t).  All 36 wavelets are padded into a
// common 384-tap frame -> one GEMM C[L x 48] = A[L x 384] * B[384 x 48],
// executed with v_wmma_f32_16x16x32_bf16 using a bf16 hi/lo split (3 WMMAs)
// for ~fp32-class accuracy on the matrix pipe.
// ---------------------------------------------------------------------------

#define L_SIG   262144      // signal length
#define PADL    192         // zero pad each side ('same' conv centering)
#define DPAD_N  (L_SIG + 512)
#define KTOT    384         // padded tap frame (12 WMMA K-steps of 32)
#define NW      48          // padded width count (36 real + 12 zero columns)
#define PI_F    3.14159265358979323846f
#define INV_PI4 0.75112554446494248f   // pi^(-1/4)

typedef __attribute__((ext_vector_type(16))) __bf16 v16bf;
typedef __attribute__((ext_vector_type(8)))  float  v8f;

union ABfrag { unsigned u32[8]; unsigned short us[16]; v16bf v; };
union Cfrag  { v8f v; float f[8]; };

__device__ __forceinline__ unsigned short f2bf(float f) {
  // round-to-nearest-even fp32 -> bf16 (finite inputs only)
  unsigned u = __float_as_uint(f);
  unsigned r = u + 0x7FFFu + ((u >> 16) & 1u);
  return (unsigned short)(r >> 16);
}
__device__ __forceinline__ float bf2f(unsigned short b) {
  return __uint_as_float(((unsigned)b) << 16);
}

// -------------------------- kernel 0: zero accumulators ---------------------
__global__ void k_zero_acc(float* acc) { acc[0] = 0.0f; acc[1] = 0.0f; }

// -------------------------- kernel 1: build tap matrix ----------------------
// Tt[n][k], n in [0,48), k in [0,384): reversed Morlet tap for width w=n+1
// at frame position k (signal index = i + k - 192).  Columns n>=36 are zero.
__global__ void k_build_T(unsigned short* __restrict__ TtH,
                          unsigned short* __restrict__ TtL) {
  int idx = blockIdx.x * blockDim.x + threadIdx.x;      // < 48*384 exactly
  int n = idx / KTOT;
  int k = idx - n * KTOT;
  float val = 0.0f;
  if (n < 36) {
    int w = n + 1;
    int N = 10 * w;
    int j = k - PADL + 5 * w;                           // index into reversed kernel
    if (j >= 0 && j < N) {
      // reversed kernel: x_j = 2*pi - 4*pi*j/(N-1)
      float x = 2.0f * PI_F - 4.0f * PI_F * (float)j / (float)(N - 1);
      float wf = (float)w;
      val = (cosf(wf * x) - expf(-0.5f * wf * wf)) * expf(-0.5f * x * x) * INV_PI4;
    }
  }
  unsigned short hv = f2bf(val);
  TtH[idx] = hv;
  TtL[idx] = f2bf(val - bf2f(hv));
}

// -------------------------- kernel 2: diff + pad + L1(wave) -----------------
__global__ void k_diff(const float* __restrict__ o, const float* __restrict__ t,
                       unsigned short* __restrict__ dH, unsigned short* __restrict__ dL,
                       float* __restrict__ acc) {
  __shared__ float blksum;
  int tid = threadIdx.x;
  if (tid == 0) blksum = 0.0f;
  __syncthreads();

  int idx = blockIdx.x * 256 + tid;                     // covers DPAD_N exactly
  float s = 0.0f;
  if (idx < DPAD_N) {
    float d = 0.0f;
    int i = idx - PADL;
    if (i >= 0 && i < L_SIG) { d = o[i] - t[i]; s = fabsf(d); }
    unsigned short hv = f2bf(d);
    dH[idx] = hv;                                       // bf16 hi
    dL[idx] = f2bf(d - bf2f(hv));                       // bf16 lo (residual)
  }
  atomicAdd(&blksum, s);                                // ds_add_f32
  __syncthreads();
  if (tid == 0) atomicAdd(acc + 0, blksum);
}

// -------------------------- kernel 3: WMMA CWT + L1(cwt) --------------------
// Block = 256 threads = 8 wave32.  Block covers 128 output positions;
// each wave owns a 16(M-outputs) x 48(N-widths) tile, K = 384 taps.
__global__ void k_cwt(const unsigned short* __restrict__ dpadH,
                      const unsigned short* __restrict__ dpadL,
                      const unsigned short* __restrict__ TtH,
                      const unsigned short* __restrict__ TtL,
                      float* __restrict__ acc) {
  __shared__ unsigned short winH[512];
  __shared__ unsigned short winL[512];
  __shared__ float blksum;

  const int tid  = threadIdx.x;
  const int lane = tid & 31;                            // wave32
  const int wv   = tid >> 5;

  // Stage the block's signal window (512 bf16 per array) via coalesced dwords.
  {
    const unsigned* gH = (const unsigned*)dpadH;
    const unsigned* gL = (const unsigned*)dpadL;
    unsigned gbase = (unsigned)blockIdx.x * 64u;        // dword units (=128 elems)
    ((unsigned*)winH)[tid] = gH[gbase + tid];
    ((unsigned*)winL)[tid] = gL[gbase + tid];
  }
  if (tid == 0) {
    blksum = 0.0f;
    // hint next block's window toward the caches (global_prefetch_b8)
    __builtin_prefetch(dpadH + ((unsigned)blockIdx.x + 1u) * 128u, 0, 1);
  }
  __syncthreads();

  const int m = lane & 15;
  const int h = lane >> 4;
  const unsigned abase = (unsigned)(wv * 16 + m);

  Cfrag accf[3];
  v8f zero = {0.f, 0.f, 0.f, 0.f, 0.f, 0.f, 0.f, 0.f};
  accf[0].v = zero; accf[1].v = zero; accf[2].v = zero;

  const unsigned* TH32 = (const unsigned*)TtH;
  const unsigned* TL32 = (const unsigned*)TtL;

  for (int kt = 0; kt < 12; ++kt) {
    const int kbase = kt * 32;

    // A fragment (16x32 bf16, contiguous layout: VGPR v -> K=2(v)+16h)
    ABfrag aH, aL;
#pragma unroll
    for (int v = 0; v < 8; ++v) {
      const int koff = 16 * h + 2 * v;
      const unsigned s = abase + (unsigned)(kbase + koff);
      aH.us[2 * v]     = winH[s];
      aH.us[2 * v + 1] = winH[s + 1];
      aL.us[2 * v]     = winL[s];
      aL.us[2 * v + 1] = winL[s + 1];
    }

#pragma unroll
    for (int nt = 0; nt < 3; ++nt) {
      // B fragment (32x16 bf16): lane holds width row n, K pairs 2v+16h
      const int n = nt * 16 + (lane & 15);
      const unsigned rb = ((unsigned)n * KTOT + (unsigned)(kbase + 16 * h)) >> 1;
      ABfrag bH, bL;
#pragma unroll
      for (int v = 0; v < 8; ++v) { bH.u32[v] = TH32[rb + v]; bL.u32[v] = TL32[rb + v]; }

      // bf16x3 split: (Ah+Al)(Bh+Bl) ~= AhBh + AhBl + AlBh  (f32 accumulate)
      accf[nt].v = __builtin_amdgcn_wmma_f32_16x16x32_bf16(
          false, aH.v, false, bH.v, (short)0, accf[nt].v, false, false);
      accf[nt].v = __builtin_amdgcn_wmma_f32_16x16x32_bf16(
          false, aH.v, false, bL.v, (short)0, accf[nt].v, false, false);
      accf[nt].v = __builtin_amdgcn_wmma_f32_16x16x32_bf16(
          false, aL.v, false, bH.v, (short)0, accf[nt].v, false, false);
    }
  }

  // |.| sum of the whole tile (zero-padded widths contribute exactly 0)
  float s = 0.0f;
#pragma unroll
  for (int r = 0; r < 8; ++r)
    s += fabsf(accf[0].f[r]) + fabsf(accf[1].f[r]) + fabsf(accf[2].f[r]);

  atomicAdd(&blksum, s);                                // ds_add_f32
  __syncthreads();
  if (tid == 0) atomicAdd(acc + 1, blksum);
}

// -------------------------- kernel 4: combine -------------------------------
__global__ void k_final(const float* __restrict__ acc, float* __restrict__ out) {
  const float invL = 1.0f / (float)L_SIG;
  out[0] = 0.5f * acc[0] * invL + 0.5f * acc[1] * (invL / 36.0f);
}

// ---------------------------------------------------------------------------
extern "C" void kernel_launch(void* const* d_in, const int* in_sizes, int n_in,
                              void* d_out, int out_size, void* d_ws, size_t ws_size,
                              hipStream_t stream) {
  (void)in_sizes; (void)n_in; (void)out_size; (void)ws_size;
  const float* o = (const float*)d_in[0];
  const float* t = (const float*)d_in[1];
  float* out = (float*)d_out;

  // workspace layout (all offsets 4B-aligned)
  char* ws = (char*)d_ws;
  float*          acc   = (float*)ws;                                  // 2 floats
  unsigned short* dpadH = (unsigned short*)(ws + 256);                 // DPAD_N u16
  unsigned short* dpadL = (unsigned short*)(ws + 256 + DPAD_N * 2);    // DPAD_N u16
  unsigned short* TtH   = (unsigned short*)(ws + 256 + DPAD_N * 4);    // NW*KTOT u16
  unsigned short* TtL   = (unsigned short*)(ws + 256 + DPAD_N * 4 + NW * KTOT * 2);

  k_zero_acc<<<1, 1, 0, stream>>>(acc);
  k_build_T<<<(NW * KTOT) / 256, 256, 0, stream>>>(TtH, TtL);          // 72 blocks
  k_diff<<<DPAD_N / 256, 256, 0, stream>>>(o, t, dpadH, dpadL, acc);   // 1026 blocks
  k_cwt<<<L_SIG / 128, 256, 0, stream>>>(dpadH, dpadL, TtH, TtL, acc); // 2048 blocks
  k_final<<<1, 1, 0, stream>>>(acc, out);
}
